// DAN_53541062312218
// MI455X (gfx1250) — matrix operand order repaired and checked by reading, not compiled, not run on hardware
//
#include <hip/hip_runtime.h>
#include <hip/hip_bf16.h>

#define VOCAB 12820
#define EMB   320
#define NROWS 8192
#define KP    12832   // VOCAB padded up to a multiple of 32

typedef __bf16 bf16_t;
typedef __attribute__((ext_vector_type(16))) __bf16 v16bf;
typedef __attribute__((ext_vector_type(8)))  float  v8f;

// ---------------------------------------------------------------------------
// Kernel 1: embeds [VOCAB][EMB] f32 (row-major)  ->  Bt [EMB][KP] bf16
// (B stored column-major along K, zero padded K in [VOCAB, KP)).
// WMMA B-fragment = one contiguous 32B read per lane.
// ---------------------------------------------------------------------------
__global__ void dan_convert_embeds(const float* __restrict__ embeds,
                                   bf16_t* __restrict__ Bt) {
    int idx = blockIdx.x * blockDim.x + threadIdx.x;
    if (idx >= EMB * KP) return;
    int e = idx / KP;
    int k = idx - e * KP;
    float v = (k < VOCAB) ? embeds[(size_t)k * EMB + e] : 0.0f;
    Bt[idx] = (bf16_t)v;
}

// ---------------------------------------------------------------------------
// Kernel 2: fused  C = x @ embeds ; rs = rowsum(x) ; out = (C/rs) @ fcw^T + b
// Block: 32 rows x 320 cols, 256 threads = 8 waves  -> 256 blocks.
// Wave (ms, nh): 16-row strip ms (0..1), col quarter nh (0..3) -> 5 N-tiles.
// Per K-step: all 5 B fragments load as one clause (immediate offsets), then
// converted-A feeds 5 back-to-back WMMAs. Row-sum is computed branch-free in
// every wave (co-executes with XDL); only the nh==0 wave publishes it.
// ---------------------------------------------------------------------------
__global__ __launch_bounds__(256)
void dan_fused_gemm(const float* __restrict__ x, const bf16_t* __restrict__ Bt,
                    const float* __restrict__ fcw, const float* __restrict__ fcb,
                    float* __restrict__ out) {
    __shared__ float s_inv[32];
    __shared__ float s_out[64];    // [32 rows][2 outputs]

    const int tid  = threadIdx.x;
    const int lane = tid & 31;
    const int w    = tid >> 5;
    const int ms   = w & 1;        // 16-row strip within 32-row block tile
    const int nh   = w >> 1;       // 0..3 -> cols [nh*80, nh*80+80)
    const int h    = lane >> 4;    // K-half selector inside fragments
    const int ln   = lane & 15;

    const int    gm   = blockIdx.x * 32 + ms * 16 + ln;   // A row of this lane
    const float* xrow = x + (size_t)gm * VOCAB;
    const int    eb   = nh * 80;

    // Per-wave moving base pointers; tile stride 16*KP*2 B fits imm24 offsets.
    const bf16_t* bbase = Bt + (size_t)(eb + ln) * KP + 16 * h;
    const float*  abase = xrow + 16 * h;

    v8f acc[5];
#pragma unroll
    for (int t = 0; t < 5; ++t) { v8f z = {}; acc[t] = z; }

    float rs = 0.0f;   // partial row-sum; computed in every wave (branch-free)

#pragma unroll 1
    for (int kk = 0; kk < 400; ++kk) {
        const int kb = kk * 32;
        __builtin_prefetch(xrow + kb + 512, 0, 0);   // ~16 iters ahead

        // ---- issue ALL B-fragment loads for this step as one clause ----
        const bf16_t* pb = bbase + kb;
        v16bf bfr[5];
#pragma unroll
        for (int t = 0; t < 5; ++t)
            bfr[t] = *(const v16bf*)(pb + (size_t)t * 16 * KP);

        // ---- A fragment: rows gm, K = kb + {16h..16h+15} ----
        const float* pa = abase + kb;
        const float4 a0 = *(const float4*)(pa + 0);
        const float4 a1 = *(const float4*)(pa + 4);
        const float4 a2 = *(const float4*)(pa + 8);
        const float4 a3 = *(const float4*)(pa + 12);

        rs += (a0.x + a0.y + a0.z + a0.w) + (a1.x + a1.y + a1.z + a1.w)
            + (a2.x + a2.y + a2.z + a2.w) + (a3.x + a3.y + a3.z + a3.w);

        v16bf a;
        a[0]  = (bf16_t)a0.x; a[1]  = (bf16_t)a0.y; a[2]  = (bf16_t)a0.z; a[3]  = (bf16_t)a0.w;
        a[4]  = (bf16_t)a1.x; a[5]  = (bf16_t)a1.y; a[6]  = (bf16_t)a1.z; a[7]  = (bf16_t)a1.w;
        a[8]  = (bf16_t)a2.x; a[9]  = (bf16_t)a2.y; a[10] = (bf16_t)a2.z; a[11] = (bf16_t)a2.w;
        a[12] = (bf16_t)a3.x; a[13] = (bf16_t)a3.y; a[14] = (bf16_t)a3.z; a[15] = (bf16_t)a3.w;

#pragma unroll
        for (int t = 0; t < 5; ++t) {
            acc[t] = __builtin_amdgcn_wmma_f32_16x16x32_bf16(
                false, a, false, bfr[t], (short)0, acc[t], false, false);
        }
    }

    // ---- K tail: kb = 12800, only 20 valid columns of x ----
    {
        const int kb = 12800;
        float av[16];
#pragma unroll
        for (int i = 0; i < 8; ++i) {
            int k0 = kb + 16 * h + i;
            int k1 = kb + 16 * h + 8 + i;
            av[i]     = (k0 < VOCAB) ? xrow[k0] : 0.0f;
            av[8 + i] = (k1 < VOCAB) ? xrow[k1] : 0.0f;
        }
#pragma unroll
        for (int i = 0; i < 16; ++i) rs += av[i];

        v16bf a;
#pragma unroll
        for (int i = 0; i < 16; ++i) a[i] = (bf16_t)av[i];

        const bf16_t* pb = bbase + kb;
#pragma unroll
        for (int t = 0; t < 5; ++t) {
            v16bf b = *(const v16bf*)(pb + (size_t)t * 16 * KP);  // zero padded
            acc[t] = __builtin_amdgcn_wmma_f32_16x16x32_bf16(
                false, a, false, b, (short)0, acc[t], false, false);
        }
    }

    // ---- rowsum reduce: lanes m and m+16 jointly cover all K of row m ----
    rs += __shfl_xor(rs, 16, 32);
    if (nh == 0 && lane < 16) s_inv[ms * 16 + lane] = 1.0f / rs;
    if (tid < 64) s_out[tid] = 0.0f;
    __syncthreads();

    // ---- fused projection to 2 outputs ----
    // C/D layout: lane holds column n=ln; VGPR r holds rows r + 8h.
    float p0[8], p1[8];
#pragma unroll
    for (int r = 0; r < 8; ++r) { p0[r] = 0.0f; p1[r] = 0.0f; }
#pragma unroll
    for (int t = 0; t < 5; ++t) {
        int   e  = eb + t * 16 + ln;
        float w0 = fcw[e];
        float w1 = fcw[EMB + e];
#pragma unroll
        for (int r = 0; r < 8; ++r) {
            float v = acc[t][r];
            p0[r] += v * w0;
            p1[r] += v * w1;
        }
    }
#pragma unroll
    for (int r = 0; r < 8; ++r) {
        int   ml  = ms * 16 + 8 * h + r;
        float inv = s_inv[ml];
        atomicAdd(&s_out[ml * 2 + 0], p0[r] * inv);
        atomicAdd(&s_out[ml * 2 + 1], p1[r] * inv);
    }
    __syncthreads();

    if (tid < 64) {
        int row = tid >> 1;
        int j   = tid & 1;
        out[(size_t)(blockIdx.x * 32 + row) * 2 + j] = s_out[tid] + fcb[j];
    }
}

// ---------------------------------------------------------------------------
extern "C" void kernel_launch(void* const* d_in, const int* in_sizes, int n_in,
                              void* d_out, int out_size, void* d_ws, size_t ws_size,
                              hipStream_t stream) {
    const float* x      = (const float*)d_in[0];   // [8192, 12820]
    const float* embeds = (const float*)d_in[1];   // [12820, 320]
    const float* fcw    = (const float*)d_in[2];   // [2, 320]
    const float* fcb    = (const float*)d_in[3];   // [2]
    float*       out    = (float*)d_out;           // [8192, 2]

    bf16_t* Bt = (bf16_t*)d_ws;                    // EMB*KP*2 = ~8.2 MB

    const int n1 = EMB * KP;
    dan_convert_embeds<<<(n1 + 255) / 256, 256, 0, stream>>>(embeds, Bt);
    dan_fused_gemm<<<NROWS / 32, 256, 0, stream>>>(x, Bt, fcw, fcb, out);
}
